// MoE_12189117186217
// MI455X (gfx1250) — matrix operand order repaired and checked by reading, not compiled, not run on hardware
//
#include <hip/hip_runtime.h>
#include <hip/hip_bf16.h>

// ---------------------------------------------------------------------------
// MoE FFN for MI455X (gfx1250): top-2 routed experts, bf16 WMMA.
// Weights are repacked per-launch into WMMA-native bf16 B-fragment tiles so
// each 16x16x32 matrix op is fed by two coalesced global_load_b128 (B) and
// two ds_load_b128 (A). Hidden activations never leave LDS. All hot global
// accesses use explicit address_space(1) so they lower to global_* (LOADcnt
// only) instead of flat_* (which would also tie up DScnt / the LDS path).
// ---------------------------------------------------------------------------

typedef __attribute__((ext_vector_type(16))) __bf16       v16bf;
typedef __attribute__((ext_vector_type(8)))  __bf16       v8bf;
typedef __attribute__((ext_vector_type(8)))  float        v8f;
typedef __attribute__((ext_vector_type(4)))  unsigned int v4u;

#define GAS __attribute__((address_space(1)))
typedef const GAS __bf16 gbf16_c;
typedef const GAS v16bf  gv16bf_c;
typedef const GAS float  gfloat_c;
typedef GAS v4u          gv4u_m;

#define WMMA_BF16(a, b, c) \
  __builtin_amdgcn_wmma_f32_16x16x32_bf16(false, (a), false, (b), (short)0, (c), false, false)

static constexpr int D    = 1024;   // model dim
static constexpr int E    = 8;      // experts
static constexpr int FF   = 2048;   // ff dim
static constexpr int NTOK = 4096;   // B*S
static constexpr int TM   = 32;     // tokens per block tile
static constexpr int FC   = 128;    // F-chunk width

// LDS row pitches: multiples of 8 bf16 (16B) for ds_load_b128, odd dword
// strides for bank-conflict freedom.
static constexpr int XLD  = D + 8;      // 1032 bf16 / row (2064B = 16*129)
static constexpr int GULD = 2 * FC + 1; // 257 f32 / row
static constexpr int HLD  = FC + 8;     // 136 bf16 / row (272B = 16*17)

static constexpr size_t XS_BYTES = (size_t)TM * XLD * 2;   // 66048
static constexpr size_t GU_BYTES = (size_t)TM * GULD * 4;  // 32896
static constexpr size_t HS_BYTES = (size_t)TM * HLD * 2;   // 8704
static constexpr size_t SMEM_BYTES = XS_BYTES + GU_BYTES + HS_BYTES + TM * 8;

// Packed weight geometry: B-tile (32K x 16N) = 32 lanes x 32B = 1KB,
// laid out [expert][ntile][ktile][lane], K-tiles contiguous.
static constexpr int GU_NT = (2 * FF) / 16;  // 256
static constexpr int GU_KT = D / 32;         // 32
static constexpr int DN_NT = D / 16;         // 64
static constexpr int DN_KT = FF / 32;        // 64
static constexpr size_t PGU_ELEMS = (size_t)E * GU_NT * GU_KT * 512;  // 33.5M
static constexpr size_t PDN_ELEMS = (size_t)E * DN_NT * DN_KT * 512;  // 16.8M

__device__ __forceinline__ gfloat_c* gfp(const float* p) {
  return (gfloat_c*)(unsigned long long)p;
}
__device__ __forceinline__ gbf16_c* gbp(const __bf16* p) {
  return (gbf16_c*)(unsigned long long)p;
}

__device__ __forceinline__ unsigned pack_bf16(float a, float b) {
  union { __bf16 h; unsigned short u; } x, y;
  x.h = (__bf16)a;
  y.h = (__bf16)b;
  return (unsigned)x.u | ((unsigned)y.u << 16);
}

// ----- A fragment (16x32 bf16, M x K) from LDS: two ds_load_b128 -----------
// lane m/m+16 holds row m; elems 0..15 at k = half*16 + 0..15
// (two 16B-contiguous loads), matching the B-fragment k order.
__device__ __forceinline__ v16bf load_a_lds(const __bf16* base, int ld, int m0,
                                            int k0, int lane) {
  const int m    = m0 + (lane & 15);
  const int half = lane >> 4;
  const __bf16* row = base + (size_t)m * ld + k0 + half * 16;
  const v8bf lo = *(const v8bf*)row;
  const v8bf hi = *(const v8bf*)(row + 8);
  return __builtin_shufflevector(lo, hi, 0, 1, 2, 3, 4, 5, 6, 7, 8, 9, 10, 11,
                                 12, 13, 14, 15);
}

// ----- fallback B fragment: scalar gather from row-major f32 global --------
__device__ __forceinline__ v16bf load_b_glb(const float* W, int ldw, int k0,
                                            int col0, int lane) {
  const int n    = lane & 15;
  const int half = lane >> 4;
  gfloat_c* p = gfp(W + (size_t)(k0 + half * 16) * ldw + col0 + n);
  v16bf b;
#pragma unroll
  for (int j = 0; j < 8; ++j) {
    b[2 * j]     = (__bf16)p[(size_t)(2 * j) * ldw];
    b[2 * j + 1] = (__bf16)p[(size_t)(2 * j + 1) * ldw];
  }
  return b;
}

// ---------------------------------------------------------------------------
// Repack: W (E x K x N row-major f32) -> bf16 B-fragment tiles.
// One thread per (tile, lane): 16 strided f32 reads, one 32B packed write.
// ---------------------------------------------------------------------------
__global__ __launch_bounds__(256) void repack_b_kernel(
    const float* __restrict__ W, __bf16* __restrict__ P, int K, int N) {
  const int KT = K >> 5;
  const int NT = N >> 4;
  const size_t id   = (size_t)blockIdx.x * 256 + threadIdx.x;
  const int    lane = (int)(id & 31);
  size_t tile = id >> 5;
  const int kt = (int)(tile % KT);
  tile /= KT;
  const int ntile = (int)(tile % NT);
  const int e     = (int)(tile / NT);
  const int n    = lane & 15;
  const int half = lane >> 4;

  gfloat_c* Wb = gfp(W + (size_t)e * K * N +
                     (size_t)(kt * 32 + half * 16) * N + ntile * 16 + n);
  unsigned w[8];
#pragma unroll
  for (int j = 0; j < 8; ++j)
    w[j] = pack_bf16(Wb[(size_t)(2 * j) * N], Wb[(size_t)(2 * j + 1) * N]);

  gv4u_m* q = (gv4u_m*)(unsigned long long)(P + (id << 4));
  const v4u lo = {w[0], w[1], w[2], w[3]};
  const v4u hi = {w[4], w[5], w[6], w[7]};
  q[0] = lo;
  q[1] = hi;
}

// ---------------------------------------------------------------------------
// Router: one wave32 per token; top-2 renormalized softmax; scatter
// (token, weight) into per-expert lists via atomics.
// ---------------------------------------------------------------------------
__global__ __launch_bounds__(256) void router_kernel(
    const float* __restrict__ x, const float* __restrict__ gw,
    int* __restrict__ counts, int* __restrict__ tok_list,
    float* __restrict__ w_list) {
  const int wv   = threadIdx.x >> 5;
  const int lane = threadIdx.x & 31;
  const int t    = blockIdx.x * 8 + wv;
  if (t >= NTOK) return;

  float acc[E];
#pragma unroll
  for (int e = 0; e < E; ++e) acc[e] = 0.f;

  gfloat_c* xr = gfp(x + (size_t)t * D);
  for (int d = lane; d < D; d += 32) {
    const float xv = xr[d];
    gfloat_c* g = gfp(gw) + (size_t)d * E;
#pragma unroll
    for (int e = 0; e < E; ++e) acc[e] += xv * g[e];
  }
#pragma unroll
  for (int e = 0; e < E; ++e) {
#pragma unroll
    for (int off = 16; off > 0; off >>= 1) acc[e] += __shfl_xor(acc[e], off, 32);
  }

  if (lane == 0) {
    int i0 = 0;
#pragma unroll
    for (int e = 1; e < E; ++e)
      if (acc[e] > acc[i0]) i0 = e;
    int i1 = (i0 == 0) ? 1 : 0;
#pragma unroll
    for (int e = 0; e < E; ++e)
      if (e != i0 && acc[e] > acc[i1]) i1 = e;
    const float w0 = 1.f / (1.f + __expf(acc[i1] - acc[i0]));
    const float w1 = 1.f - w0;
    int p = atomicAdd(&counts[i0], 1);
    tok_list[i0 * NTOK + p] = t;
    w_list[i0 * NTOK + p]   = w0;
    p = atomicAdd(&counts[i1], 1);
    tok_list[i1 * NTOK + p] = t;
    w_list[i1 * NTOK + p]   = w1;
  }
}

// ---------------------------------------------------------------------------
// Fused expert kernel: block = 32 tokens of one expert, full D output.
// Per 128-wide F chunk: GEMM1 (gate|up) -> SiLU*up*route_w (bf16, LDS)
// -> GEMM2 partial accumulate. 8 waves: M tile row (w&1), col group (w>>1).
// ---------------------------------------------------------------------------
template <bool PACKED>
__global__ __launch_bounds__(256) void moe_expert_kernel(
    const float* __restrict__ x, const float* __restrict__ gate_up_w,
    const float* __restrict__ down_w, const __bf16* __restrict__ pgu,
    const __bf16* __restrict__ pdn, const int* __restrict__ counts,
    const int* __restrict__ tok_list, const float* __restrict__ w_list,
    float* __restrict__ out) {
  const int e  = blockIdx.y;
  const int ne = counts[e];
  const int m0 = blockIdx.x * TM;
  if (m0 >= ne) return;  // uniform across the block: safe early exit

  extern __shared__ char smem[];
  __bf16* Xs  = (__bf16*)smem;
  float*  GUs = (float*)(smem + XS_BYTES);
  __bf16* Hs  = (__bf16*)(smem + XS_BYTES + GU_BYTES);
  int*    tok = (int*)(smem + XS_BYTES + GU_BYTES + HS_BYTES);
  float*  wts = (float*)(smem + XS_BYTES + GU_BYTES + HS_BYTES + TM * 4);

  const int tid  = threadIdx.x;
  const int wv   = tid >> 5;
  const int lane = tid & 31;
  const int mr   = wv & 1;   // M tile row (0/1)
  const int cg   = wv >> 1;  // column group (0..3)
  const int nn   = lane & 15;
  const int half = lane >> 4;

  const float* gup = gate_up_w + (size_t)e * D * (2 * FF);
  const float* wd  = down_w + (size_t)e * FF * D;

  if (tid < TM) {
    const int slot   = m0 + tid;
    const bool valid = slot < ne;
    tok[tid] = valid ? tok_list[e * NTOK + slot] : 0;
    wts[tid] = valid ? w_list[e * NTOK + slot] : 0.f;
  }
  __syncthreads();

  // stage x tile (32 x 1024) as bf16 into LDS, zero-padding invalid rows
  {
    gfloat_c* xg = gfp(x);
    for (int idx = tid; idx < TM * D; idx += 256) {
      const int r = idx >> 10;
      const int c = idx & (D - 1);
      const float v = (m0 + r < ne) ? xg[(size_t)tok[r] * D + c] : 0.f;
      Xs[(size_t)r * XLD + c] = (__bf16)v;
    }
  }
  __syncthreads();

  const v8f vzero = {0.f, 0.f, 0.f, 0.f, 0.f, 0.f, 0.f, 0.f};
  v8f acc2[16];
#pragma unroll
  for (int i = 0; i < 16; ++i) acc2[i] = vzero;

  for (int fc = 0; fc < FF / FC; ++fc) {
    // ---- GEMM1: (32 x 1024) @ (1024 x 256) -> gate|up chunk ----
    v8f acc1[4];
#pragma unroll
    for (int i = 0; i < 4; ++i) acc1[i] = vzero;

    gbf16_c* pb1[4];
    if (PACKED) {
#pragma unroll
      for (int i = 0; i < 4; ++i) {
        const int cc    = cg * 4 + i;
        const int ntile = (cc < 8) ? (fc * 8 + cc) : (128 + fc * 8 + (cc - 8));
        pb1[i] =
            gbp(pgu) + (((size_t)e * GU_NT + ntile) * GU_KT) * 512 + lane * 16;
      }
    }
    for (int kk = 0; kk < D; kk += 32) {
      const v16bf a = load_a_lds(Xs, XLD, mr * 16, kk, lane);
      if (PACKED)  // WGP-scope streaming prefetch ~4 K-tiles ahead
        __builtin_prefetch((const void*)((unsigned long long)pb1[0] + 4096), 0, 3);
#pragma unroll
      for (int i = 0; i < 4; ++i) {
        v16bf b;
        if (PACKED) {
          b = *(gv16bf_c*)pb1[i];  // two global_load_b128, coalesced 1KB/wave
          pb1[i] += 512;
        } else {
          const int cc = cg * 4 + i;
          const int h  = (cc < 8) ? (fc * FC + cc * 16)
                                  : (FF + fc * FC + (cc - 8) * 16);
          b = load_b_glb(gup, 2 * FF, kk, h, lane);
        }
        acc1[i] = WMMA_BF16(a, b, acc1[i]);
      }
    }
    __syncthreads();  // all waves done reading Hs/GUs of previous chunk

    // spill gate|up f32 tiles to LDS (C layout: VGPR v -> M = half*8 + v)
#pragma unroll
    for (int i = 0; i < 4; ++i) {
      const int cc = cg * 4 + i;
#pragma unroll
      for (int v = 0; v < 8; ++v) {
        const int m = mr * 16 + half * 8 + v;
        GUs[(size_t)m * GULD + cc * 16 + nn] = acc1[i][v];
      }
    }
    __syncthreads();

    // ---- SiLU(gate) * up * routing_weight -> bf16 hidden chunk ----
#pragma unroll 4
    for (int r = 0; r < 16; ++r) {
      const int idx = r * 256 + tid;
      const int m = idx >> 7;
      const int j = idx & (FC - 1);
      const float g = GUs[(size_t)m * GULD + j];
      const float u = GUs[(size_t)m * GULD + FC + j];
      const float s = g / (1.f + __expf(-g));
      Hs[(size_t)m * HLD + j] = (__bf16)(s * u * wts[m]);
    }
    __syncthreads();

    // ---- GEMM2 partial: (32 x 128) @ (128 x 1024) accumulated ----
    gbf16_c* pb2 = gbp(pdn) +
                   (((size_t)e * DN_NT + cg * 16) * DN_KT + fc * 4) * 512 +
                   lane * 16;
    for (int ks = 0; ks < FC / 32; ++ks) {
      const v16bf a = load_a_lds(Hs, HLD, mr * 16, ks * 32, lane);
      if (PACKED)
        __builtin_prefetch((const void*)((unsigned long long)pb2 + 4096), 0, 3);
#pragma unroll
      for (int i = 0; i < 16; ++i) {
        v16bf b;
        if (PACKED) {
          b = *(gv16bf_c*)(pb2 + (size_t)i * DN_KT * 512 + ks * 512);
        } else {
          b = load_b_glb(wd, D, fc * FC + ks * 32, (cg * 16 + i) * 16, lane);
        }
        acc2[i] = WMMA_BF16(a, b, acc2[i]);
      }
    }
  }

  // ---- scatter weighted expert output (routing weight already applied) ----
#pragma unroll
  for (int i = 0; i < 16; ++i) {
    const int d0 = (cg * 16 + i) * 16;
#pragma unroll
    for (int v = 0; v < 8; ++v) {
      const int ml = mr * 16 + half * 8 + v;
      if (m0 + ml < ne) {
        atomicAdd(&out[(size_t)tok[ml] * D + d0 + nn], acc2[i][v]);
      }
    }
  }
}

// ---------------------------------------------------------------------------
extern "C" void kernel_launch(void* const* d_in, const int* in_sizes, int n_in,
                              void* d_out, int out_size, void* d_ws,
                              size_t ws_size, hipStream_t stream) {
  const float* x         = (const float*)d_in[0];  // (2,2048,1024) f32
  const float* gate_w    = (const float*)d_in[1];  // (1024,8) f32
  const float* gate_up_w = (const float*)d_in[2];  // (8,1024,4096) f32
  const float* down_w    = (const float*)d_in[3];  // (8,2048,1024) f32
  float* out = (float*)d_out;
  (void)out;

  const size_t PGU_BYTES  = PGU_ELEMS * 2;                  // 67,108,864
  const size_t PDN_BYTES  = PDN_ELEMS * 2;                  // 33,554,432
  const size_t LIST_BYTES = 256 + (size_t)E * NTOK * 8;     // counts + lists
  const bool packed = ws_size >= PGU_BYTES + PDN_BYTES + LIST_BYTES;

  char* wsp = (char*)d_ws;
  __bf16* pgu = (__bf16*)wsp;
  __bf16* pdn = (__bf16*)(wsp + PGU_BYTES);
  char* meta  = packed ? (wsp + PGU_BYTES + PDN_BYTES) : wsp;
  int*   counts   = (int*)meta;
  int*   tok_list = (int*)(meta + 256);
  float* w_list   = (float*)(meta + 256 + (size_t)E * NTOK * sizeof(int));

  (void)hipMemsetAsync(counts, 0, E * sizeof(int), stream);
  (void)hipMemsetAsync(d_out, 0, (size_t)out_size * sizeof(float), stream);

  if (packed) {
    repack_b_kernel<<<(int)(PGU_ELEMS / 16 / 256), 256, 0, stream>>>(
        gate_up_w, pgu, D, 2 * FF);
    repack_b_kernel<<<(int)(PDN_ELEMS / 16 / 256), 256, 0, stream>>>(
        down_w, pdn, FF, D);
  }

  router_kernel<<<NTOK / 8, 256, 0, stream>>>(x, gate_w, counts, tok_list,
                                              w_list);

  dim3 grid(NTOK / TM, E);  // worst case: one expert owns all tokens
  if (packed) {
    moe_expert_kernel<true><<<grid, 256, SMEM_BYTES, stream>>>(
        x, gate_up_w, down_w, pgu, pdn, counts, tok_list, w_list, out);
  } else {
    moe_expert_kernel<false><<<grid, 256, SMEM_BYTES, stream>>>(
        x, gate_up_w, down_w, pgu, pdn, counts, tok_list, w_list, out);
  }
}
